// NaryDecoderCell8_73967926772620
// MI455X (gfx1250) — matrix operand order repaired and checked by reading, not compiled, not run on hardware
//
#include <hip/hip_runtime.h>

// Problem sizes (compile-time constants from the reference)
#define B_SZ   4096
#define H_SZ   512
#define D_DEG  8
#define C_SZ   512
#define DEP_SZ 8
#define IN_SZ  1552      // C + 2H + 8 + 8
#define IN_PAD 1600      // padded to multiple of 64 (K stage depth)
#define G3H    1536      // 3*H
#define KSTEP  64        // K elements staged per TDM tile

typedef __attribute__((ext_vector_type(16))) __bf16 v16bf;
typedef __attribute__((ext_vector_type(8)))  float  v8f;
typedef __attribute__((ext_vector_type(4)))  unsigned int u32x4;
typedef __attribute__((ext_vector_type(8)))  int  i32x8;
typedef __attribute__((ext_vector_type(4)))  int  i32x4;

union FragAB {
    v16bf v;
    uint4 q[2];
};

__device__ __forceinline__ unsigned short f32_to_bf16(float f) {
    unsigned int u = __float_as_uint(f);
    u += 0x7FFFu + ((u >> 16) & 1u);   // round to nearest even
    return (unsigned short)(u >> 16);
}

__device__ __forceinline__ float sigmoidf(float x) {
    return 1.0f / (1.0f + __expf(-x));
}

// ---------------------------------------------------------------------------
// Pack x = concat(parent_out, encoding, parent_h, pos, depth) -> bf16 [B, IN_PAD]
// ---------------------------------------------------------------------------
__global__ void pack_x_kernel(const float* __restrict__ po, const float* __restrict__ enc,
                              const float* __restrict__ ph, const float* __restrict__ pos,
                              const float* __restrict__ dep, unsigned short* __restrict__ x) {
    int idx = blockIdx.x * 256 + threadIdx.x;
    if (idx >= B_SZ * IN_PAD) return;
    int b = idx / IN_PAD, k = idx - b * IN_PAD;
    float v = 0.0f;
    if (k < C_SZ)                        v = po [(size_t)b * C_SZ + k];
    else if (k < C_SZ + H_SZ)            v = enc[(size_t)b * H_SZ + (k - C_SZ)];
    else if (k < C_SZ + 2 * H_SZ)        v = ph [(size_t)b * H_SZ + (k - C_SZ - H_SZ)];
    else if (k < C_SZ + 2 * H_SZ + 8)    v = pos[(size_t)b * 8 + (k - C_SZ - 2 * H_SZ)];
    else if (k < IN_SZ)                  v = dep[(size_t)b * 8 + (k - C_SZ - 2 * H_SZ - 8)];
    x[idx] = f32_to_bf16(v);
}

// W_ih fp32 [D,3H,IN] -> bf16 [D,3H,IN_PAD] (zero padded K)
__global__ void conv_wih_kernel(const float* __restrict__ w, unsigned short* __restrict__ o) {
    int idx = blockIdx.x * 256 + threadIdx.x;
    if (idx >= D_DEG * G3H * IN_PAD) return;
    int k   = idx % IN_PAD;
    int row = idx / IN_PAD;
    o[idx] = (k < IN_SZ) ? f32_to_bf16(w[(size_t)row * IN_SZ + k]) : (unsigned short)0;
}

__global__ void conv_f32_bf16_kernel(const float* __restrict__ src,
                                     unsigned short* __restrict__ dst, int n) {
    int idx = blockIdx.x * 256 + threadIdx.x;
    if (idx < n) dst[idx] = f32_to_bf16(src[idx]);
}

__global__ void zero_h_kernel(float* __restrict__ h, unsigned short* __restrict__ h_bf) {
    int idx = blockIdx.x * 256 + threadIdx.x;
    if (idx < B_SZ * H_SZ) { h[idx] = 0.0f; h_bf[idx] = 0; }
}

// ---------------------------------------------------------------------------
// bf16 GEMM: C[M,N](fp32, row-major) (+)= A[M,K](bf16 row-major) * Bw[N,K]^T
//
// Block: 256 threads = 8 waves, tile 128(M) x 256(N); wave tile 64x64
// (16 accumulators of v_wmma_f32_16x16x32_bf16).
// K staged in 64-deep double-buffered LDS tiles filled by the Tensor Data
// Mover (A: 128x64 = 16KB, B: 256x64 = 32KB; 96KB dynamic LDS total ->
// 3 blocks per 320KB WGP so TDM/barrier latency overlaps other blocks).
// Per stage, per wave: 32 ds_load_b128 feed 32 WMMAs between two barriers.
// M % 128 == 0, N % 256 == 0, K % 64 == 0 assumed.
// ---------------------------------------------------------------------------
#if __has_builtin(__builtin_amdgcn_tensor_load_to_lds)
#define USE_TDM 1
#endif

#ifdef USE_TDM
// 2-D TDM load: tile_dim0 = KSTEP elems (2B each), tile_dim1 = rows,
// row stride = stride_elems (elements).
__device__ __forceinline__ void tdm_load_tile(unsigned int lds_byte_addr,
                                              const unsigned short* gaddr,
                                              unsigned int rows,
                                              unsigned long long stride_elems) {
    unsigned long long ga = (unsigned long long)(const void*)gaddr;
    u32x4 g0;
    g0[0] = 1u;                                   // count=1, is_restore=0, gather off
    g0[1] = lds_byte_addr;                        // lds_addr [63:32]
    g0[2] = (unsigned int)(ga & 0xFFFFFFFFu);     // global_addr[31:0]
    g0[3] = (unsigned int)((ga >> 32) & 0x01FFFFFFu) | (2u << 30); // addr[56:32], type=2
    i32x8 g1;
    g1[0] = (int)(1u << 16);                      // wg_mask=0, data_size=1 (2B)
    g1[1] = (int)((unsigned)KSTEP << 16);         // tensor_dim0[15:0] (bits 63:48)
    g1[2] = (int)(rows << 16);                    // tensor_dim0[31:16]=0, tensor_dim1[15:0]
    g1[3] = (int)((unsigned)KSTEP << 16);         // tensor_dim1[31:16]=0, tile_dim0
    g1[4] = (int)(rows);                          // tile_dim1, tile_dim2=0
    g1[5] = (int)(stride_elems & 0xFFFFFFFFu);    // tensor_dim0_stride[31:0]
    g1[6] = (int)((stride_elems >> 32) & 0xFFFFu);// dim0_stride[47:32], dim1_stride lo = 0
    g1[7] = 0;
    i32x4 z4 = {0, 0, 0, 0};
#if defined(__clang_major__) && (__clang_major__ >= 23)
    i32x8 z8 = {0, 0, 0, 0, 0, 0, 0, 0};
    __builtin_amdgcn_tensor_load_to_lds(g0, g1, z4, z4, z8, 0);
#else
    __builtin_amdgcn_tensor_load_to_lds(g0, g1, z4, z4, 0);
#endif
}
#endif

#define LDS_A_ELEMS (128 * KSTEP)   // per buffer
#define LDS_B_ELEMS (256 * KSTEP)   // per buffer
#define GEMM_LDS_BYTES ((2 * LDS_A_ELEMS + 2 * LDS_B_ELEMS) * 2)

__global__ __launch_bounds__(256) void gemm_bf16_wmma(
        const unsigned short* __restrict__ A, int lda,
        const unsigned short* __restrict__ Bw, int ldb,
        float* __restrict__ C, int ldc, int K, int accumulate) {
    const int lane = threadIdx.x & 31;
    const int wave = threadIdx.x >> 5;
    const int wm = wave & 1;        // 2 wave rows  -> 64 M each
    const int wn = wave >> 1;       // 4 wave cols  -> 64 N each
    const int row0 = blockIdx.x * 128 + wm * 64;
    const int col0 = blockIdx.y * 256 + wn * 64;
    const int lo = lane & 15;
    const int hi = lane >> 4;

    v8f acc[4][4];
#pragma unroll
    for (int mi = 0; mi < 4; ++mi)
#pragma unroll
        for (int ni = 0; ni < 4; ++ni) {
            if (accumulate) {
#pragma unroll
                for (int v = 0; v < 8; ++v)
                    acc[mi][ni][v] =
                        C[(size_t)(row0 + mi * 16 + hi * 8 + v) * ldc + (col0 + ni * 16 + lo)];
            } else {
#pragma unroll
                for (int v = 0; v < 8; ++v) acc[mi][ni][v] = 0.0f;
            }
        }

#ifdef USE_TDM
    extern __shared__ unsigned short lds[];
    unsigned short* ldsA = lds;                       // [2][128*KSTEP]
    unsigned short* ldsB = lds + 2 * LDS_A_ELEMS;     // [2][256*KSTEP]

    const unsigned short* a_tile0 = A + (size_t)(blockIdx.x * 128) * lda;
    const unsigned short* b_tile0 = Bw + (size_t)(blockIdx.y * 256) * ldb;
    const int nk = K / KSTEP;

    if (wave == 0) {
        tdm_load_tile((unsigned int)(size_t)&ldsA[0], a_tile0, 128u,
                      (unsigned long long)lda);
        tdm_load_tile((unsigned int)(size_t)&ldsB[0], b_tile0, 256u,
                      (unsigned long long)ldb);
    }

    for (int i = 0; i < nk; ++i) {
        const int cur = i & 1;
        if (wave == 0) {
            if (i + 1 < nk) {
                // previous end-of-step barrier guarantees buf[1-cur] is free
                tdm_load_tile((unsigned int)(size_t)&ldsA[(1 - cur) * LDS_A_ELEMS],
                              a_tile0 + (size_t)(i + 1) * KSTEP, 128u,
                              (unsigned long long)lda);
                tdm_load_tile((unsigned int)(size_t)&ldsB[(1 - cur) * LDS_B_ELEMS],
                              b_tile0 + (size_t)(i + 1) * KSTEP, 256u,
                              (unsigned long long)ldb);
                __builtin_amdgcn_s_wait_tensorcnt(2);  // oldest 2 (current tile) done
            } else {
                __builtin_amdgcn_s_wait_tensorcnt(0);
            }
        }
        __syncthreads();   // publish buf[cur] to all waves

        const unsigned short* tA = &ldsA[cur * LDS_A_ELEMS];
        const unsigned short* tB = &ldsB[cur * LDS_B_ELEMS];
#pragma unroll
        for (int kc = 0; kc < KSTEP / 32; ++kc) {
            FragAB af[4];
#pragma unroll
            for (int mi = 0; mi < 4; ++mi) {
                const uint4* p =
                    (const uint4*)&tA[(wm * 64 + mi * 16 + lo) * KSTEP + kc * 32 + hi * 16];
                af[mi].q[0] = p[0];   // K = kc*32 + hi*16 .. +7
                af[mi].q[1] = p[1];   // K = kc*32 + hi*16 + 8 .. +15
            }
            FragAB bfr[4];
#pragma unroll
            for (int ni = 0; ni < 4; ++ni) {
                const uint4* p =
                    (const uint4*)&tB[(wn * 64 + ni * 16 + lo) * KSTEP + kc * 32 + hi * 16];
                bfr[ni].q[0] = p[0];
                bfr[ni].q[1] = p[1];
            }
#pragma unroll
            for (int mi = 0; mi < 4; ++mi)
#pragma unroll
                for (int ni = 0; ni < 4; ++ni)
                    acc[mi][ni] = __builtin_amdgcn_wmma_f32_16x16x32_bf16(
                        false, af[mi].v, false, bfr[ni].v, (short)0, acc[mi][ni],
                        false, false);
        }
        __syncthreads();   // all reads of buf[cur] done before it is refilled
    }
#else
    // Fallback: direct global-load fragments (proven codegen from round 1).
    const unsigned short* a_base = A + (size_t)(row0 + lo) * lda + hi * 16;
    const unsigned short* b_base = Bw + (size_t)(col0 + lo) * ldb + hi * 16;

    for (int k0 = 0; k0 < K; k0 += 32) {
        FragAB af[4];
#pragma unroll
        for (int mi = 0; mi < 4; ++mi) {
            const uint4* p = (const uint4*)(a_base + (size_t)mi * 16 * lda + k0);
            af[mi].q[0] = p[0];
            af[mi].q[1] = p[1];
        }
        FragAB bfr[4];
#pragma unroll
        for (int ni = 0; ni < 4; ++ni) {
            const uint4* p = (const uint4*)(b_base + (size_t)ni * 16 * ldb + k0);
            bfr[ni].q[0] = p[0];
            bfr[ni].q[1] = p[1];
        }
#pragma unroll
        for (int mi = 0; mi < 4; ++mi)
#pragma unroll
            for (int ni = 0; ni < 4; ++ni)
                acc[mi][ni] = __builtin_amdgcn_wmma_f32_16x16x32_bf16(
                    false, af[mi].v, false, bfr[ni].v, (short)0, acc[mi][ni], false, false);
    }
#endif

#pragma unroll
    for (int mi = 0; mi < 4; ++mi)
#pragma unroll
        for (int ni = 0; ni < 4; ++ni)
#pragma unroll
            for (int v = 0; v < 8; ++v)
                C[(size_t)(row0 + mi * 16 + hi * 8 + v) * ldc + (col0 + ni * 16 + lo)] =
                    acc[mi][ni][v];
}

// ---------------------------------------------------------------------------
// GRU gate elementwise for one degree: adds biases, updates h in place (fp32)
// and writes bf16 copy for the next GEMMs.
// ---------------------------------------------------------------------------
__global__ void gru_elem_kernel(const float* __restrict__ gi, const float* __restrict__ gh,
                                const float* __restrict__ bih, const float* __restrict__ bhh,
                                float* __restrict__ h, unsigned short* __restrict__ h_bf) {
    int idx = blockIdx.x * 256 + threadIdx.x;
    if (idx >= B_SZ * H_SZ) return;
    int b = idx / H_SZ, j = idx - b * H_SZ;
    size_t g = (size_t)b * G3H;
    float ir = gi[g + j]             + bih[j];
    float iz = gi[g + H_SZ + j]      + bih[H_SZ + j];
    float in = gi[g + 2 * H_SZ + j]  + bih[2 * H_SZ + j];
    float hr = gh[g + j]             + bhh[j];
    float hz = gh[g + H_SZ + j]      + bhh[H_SZ + j];
    float hn = gh[g + 2 * H_SZ + j]  + bhh[2 * H_SZ + j];
    float r = sigmoidf(ir + hr);
    float z = sigmoidf(iz + hz);
    float n = tanhf(in + r * hn);
    float hv = (1.0f - z) * n + z * h[idx];
    h[idx]    = hv;
    h_bf[idx] = f32_to_bf16(hv);
}

// ---------------------------------------------------------------------------
// prob[b,d] = sigmoid(h[b,:] . Wp[d,:] + bp[d]); one wave per batch row.
// ---------------------------------------------------------------------------
__global__ __launch_bounds__(256) void prob_kernel(const float* __restrict__ h,
                                                   const float* __restrict__ Wp_d,
                                                   const float* __restrict__ bp, int d,
                                                   float* __restrict__ probs) {
    int wave = threadIdx.x >> 5;
    int lane = threadIdx.x & 31;
    int b = blockIdx.x * 8 + wave;
    if (b >= B_SZ) return;
    float s = 0.0f;
    for (int j = lane; j < H_SZ; j += 32)
        s += h[(size_t)b * H_SZ + j] * Wp_d[j];
#pragma unroll
    for (int off = 16; off > 0; off >>= 1)
        s += __shfl_xor(s, off, 32);
    if (lane == 0)
        probs[(size_t)b * D_DEG + d] = sigmoidf(s + bp[d]);
}

// out_h = tanh(comb_sum + sum_d bh[d])
__global__ void final_tanh_kernel(const float* __restrict__ comb,
                                  const float* __restrict__ bh, float* __restrict__ out) {
    int idx = blockIdx.x * 256 + threadIdx.x;
    if (idx >= B_SZ * H_SZ) return;
    int j = idx % H_SZ;
    float s = comb[idx];
#pragma unroll
    for (int d = 0; d < D_DEG; ++d) s += bh[d * H_SZ + j];
    out[idx] = tanhf(s);
}

// ---------------------------------------------------------------------------
extern "C" void kernel_launch(void* const* d_in, const int* in_sizes, int n_in,
                              void* d_out, int out_size, void* d_ws, size_t ws_size,
                              hipStream_t stream) {
    (void)in_sizes; (void)n_in; (void)out_size; (void)ws_size;

    const float* parent_out = (const float*)d_in[0];
    const float* encoding   = (const float*)d_in[1];
    const float* parent_h   = (const float*)d_in[2];
    const float* pos        = (const float*)d_in[3];
    const float* depth      = (const float*)d_in[4];
    const float* W_ih       = (const float*)d_in[5];
    const float* W_hh       = (const float*)d_in[6];
    const float* b_ih       = (const float*)d_in[7];
    const float* b_hh       = (const float*)d_in[8];
    const float* Wp         = (const float*)d_in[9];
    const float* bp         = (const float*)d_in[10];
    const float* Wh         = (const float*)d_in[11];
    const float* bh         = (const float*)d_in[12];

    // Workspace carve-up (256B aligned blocks), ~135 MB total.
    char* ws = (char*)d_ws;
    size_t off = 0;
    auto carve = [&](size_t bytes) -> void* {
        void* p = ws + off;
        off += (bytes + 255) & ~(size_t)255;
        return p;
    };
    unsigned short* x_bf   = (unsigned short*)carve((size_t)B_SZ * IN_PAD * 2);
    unsigned short* wih_bf = (unsigned short*)carve((size_t)D_DEG * G3H * IN_PAD * 2);
    unsigned short* whh_bf = (unsigned short*)carve((size_t)D_DEG * G3H * H_SZ * 2);
    unsigned short* wh_bf  = (unsigned short*)carve((size_t)D_DEG * H_SZ * H_SZ * 2);
    unsigned short* h_bf   = (unsigned short*)carve((size_t)B_SZ * H_SZ * 2);
    float* h_f32 = (float*)carve((size_t)B_SZ * H_SZ * 4);
    float* gi    = (float*)carve((size_t)B_SZ * G3H * 4);
    float* gh    = (float*)carve((size_t)B_SZ * G3H * 4);
    float* comb  = (float*)carve((size_t)B_SZ * H_SZ * 4);

    float* out_h = (float*)d_out;
    float* out_p = out_h + (size_t)B_SZ * H_SZ;

    dim3 blk(256);

    pack_x_kernel<<<(B_SZ * IN_PAD + 255) / 256, blk, 0, stream>>>(
        parent_out, encoding, parent_h, pos, depth, x_bf);
    conv_wih_kernel<<<(D_DEG * G3H * IN_PAD + 255) / 256, blk, 0, stream>>>(W_ih, wih_bf);
    conv_f32_bf16_kernel<<<(D_DEG * G3H * H_SZ + 255) / 256, blk, 0, stream>>>(
        W_hh, whh_bf, D_DEG * G3H * H_SZ);
    conv_f32_bf16_kernel<<<(D_DEG * H_SZ * H_SZ + 255) / 256, blk, 0, stream>>>(
        Wh, wh_bf, D_DEG * H_SZ * H_SZ);
    zero_h_kernel<<<(B_SZ * H_SZ + 255) / 256, blk, 0, stream>>>(h_f32, h_bf);

    for (int d = 0; d < D_DEG; ++d) {
        // gi = x @ W_ih[d]^T   (M=4096, N=1536, K=1600)
        gemm_bf16_wmma<<<dim3(B_SZ / 128, G3H / 256), blk, GEMM_LDS_BYTES, stream>>>(
            x_bf, IN_PAD, wih_bf + (size_t)d * G3H * IN_PAD, IN_PAD, gi, G3H, IN_PAD, 0);
        // gh = h @ W_hh[d]^T   (M=4096, N=1536, K=512) using h from previous step
        gemm_bf16_wmma<<<dim3(B_SZ / 128, G3H / 256), blk, GEMM_LDS_BYTES, stream>>>(
            h_bf, H_SZ, whh_bf + (size_t)d * G3H * H_SZ, H_SZ, gh, G3H, H_SZ, 0);
        // GRU gates -> h_new (fp32 in place + bf16 copy)
        gru_elem_kernel<<<(B_SZ * H_SZ + 255) / 256, blk, 0, stream>>>(
            gi, gh, b_ih + (size_t)d * G3H, b_hh + (size_t)d * G3H, h_f32, h_bf);
        // comb_sum (+)= h_new @ Wh[d]^T  (M=4096, N=512, K=512)
        gemm_bf16_wmma<<<dim3(B_SZ / 128, H_SZ / 256), blk, GEMM_LDS_BYTES, stream>>>(
            h_bf, H_SZ, wh_bf + (size_t)d * H_SZ * H_SZ, H_SZ, comb, H_SZ, H_SZ, d > 0);
        // probs[:, d] = sigmoid(h_new @ Wp[d] + bp[d])
        prob_kernel<<<B_SZ / 8, blk, 0, stream>>>(h_f32, Wp + (size_t)d * H_SZ, bp, d, out_p);
    }

    final_tanh_kernel<<<(B_SZ * H_SZ + 255) / 256, blk, 0, stream>>>(comb, bh, out_h);
}
